// ParticleFilter_40810779247472
// MI455X (gfx1250) — matrix operand order repaired and checked by reading, not compiled, not run on hardware
//
#include <hip/hip_runtime.h>
#include <hip/hip_bf16.h>

typedef __attribute__((ext_vector_type(16))) __bf16 v16bf;
typedef __attribute__((ext_vector_type(8)))  __bf16 v8bf;
typedef __attribute__((ext_vector_type(8)))  float  v8f;

#define PF_B    32
#define PF_T    64
#define PF_P    2048
#define PF_D    32
#define PF_DOBS 16
#define NTHREADS 512           // 16 waves of 32
#define NWAVES   (NTHREADS / 32)
#define LOG_EPS  (-50.0f)

__device__ __forceinline__ unsigned hash3(unsigned a, unsigned b, unsigned c) {
    unsigned h = a * 0x9E3779B1u ^ b * 0x85EBCA77u ^ c * 0xC2B2AE3Du;
    h ^= h >> 16; h *= 0x7FEB352Du;
    h ^= h >> 15; h *= 0x846CA68Bu;
    h ^= h >> 16;
    return h;
}

__device__ __forceinline__ float uniform01(unsigned h) {
    return (float)(h >> 8) * (1.0f / 16777216.0f);
}

__device__ __forceinline__ float gauss(unsigned a, unsigned b, unsigned c) {
    float u1 = uniform01(hash3(a, b, c)) + 1e-7f;
    float u2 = uniform01(hash3(a ^ 0x5BF03635u, b, c));
    return __builtin_sqrtf(-2.0f * __logf(u1)) * __cosf(6.28318530718f * u2);
}

__launch_bounds__(NTHREADS)
__global__ void pf_persistent_kernel(const float* __restrict__ obs,        // [B,T,DOBS]
                                     const float* __restrict__ init_p,     // [B,P,D]
                                     const float* __restrict__ Amat,       // [D,D]
                                     const float* __restrict__ bvec,       // [D]
                                     const float* __restrict__ log_sigma,  // [D]
                                     const float* __restrict__ Cmat,       // [DOBS,D]
                                     const float* __restrict__ log_r,      // [DOBS]
                                     float* __restrict__ pseq,             // [B,T,P,D]
                                     float* __restrict__ wseq,             // [B,T,P]
                                     __bf16* __restrict__ ws)              // 2*B*P*D bf16 state
{
    __shared__ float          s_w[PF_P];       // log-weights -> CDF
    __shared__ unsigned short s_idx[PF_P];     // resampling indices
    __shared__ float          s_wave[NWAVES + 1];
    __shared__ float          s_red[NWAVES];

    const int b    = blockIdx.x;
    const int tid  = threadIdx.x;
    const int lane = tid & 31;
    const int wave = tid >> 5;
    const int nl   = lane & 15;        // column within 16-wide tile
    const int halfv = lane >> 4;       // 0 / 1 half of wave
    const int kb   = halfv << 4;       // unused (0 or 16)
    const int koff = halfv << 3;       // A-fragment K base (0 or 8)

    // ---- time-invariant B fragments (Aᵀ as two N-tiles, Cᵀ) ----
    // B[k][n] = A[n][k] (forward) / C[n][k] (measurement); element q <-> k = koff + q + 8*(q/8)
    v16bf bA0, bA1, bC;
#pragma unroll
    for (int q = 0; q < 16; ++q) {
        bA0[q] = (__bf16)Amat[(nl)      * PF_D + koff + q + ((q >> 3) << 3)];
        bA1[q] = (__bf16)Amat[(16 + nl) * PF_D + koff + q + ((q >> 3) << 3)];
        bC[q]  = (__bf16)Cmat[(nl)      * PF_D + koff + q + ((q >> 3) << 3)];
    }
    const float bv0 = bvec[nl];
    const float bv1 = bvec[16 + nl];
    const float sg0 = __expf(log_sigma[nl]);
    const float sg1 = __expf(log_sigma[16 + nl]);
    const float rinv = __expf(-log_r[nl]);

    float sum_log_r = 0.0f;
#pragma unroll
    for (int o = 0; o < PF_DOBS; ++o) sum_log_r += log_r[o];
    const float const_term = -sum_log_r - 0.5f * (float)PF_DOBS * 1.8378770664093453f; // log(2*pi)

    __bf16* buf0 = ws + (size_t)b * PF_P * PF_D;
    __bf16* buf1 = ws + (size_t)PF_B * PF_P * PF_D + (size_t)b * PF_P * PF_D;

    // initial log-weights: uniform (0 == log(1/P) up to the normalizing constant)
    for (int i = tid; i < PF_P; i += NTHREADS) s_w[i] = 0.0f;
    __syncthreads();

    int cur = 0;
    for (int t = 0; t < PF_T; ++t) {
        // ================= Phase 1: normalize + CDF + resample =================
        const int i0 = tid * 4;
        float w0 = s_w[i0 + 0], w1 = s_w[i0 + 1], w2 = s_w[i0 + 2], w3 = s_w[i0 + 3];

        // block max
        float m = fmaxf(fmaxf(w0, w1), fmaxf(w2, w3));
#pragma unroll
        for (int off = 16; off >= 1; off >>= 1) m = fmaxf(m, __shfl_xor(m, off));
        if (lane == 0) s_red[wave] = m;
        __syncthreads();
        if (tid == 0) {
            float mm = s_red[0];
#pragma unroll
            for (int i = 1; i < NWAVES; ++i) mm = fmaxf(mm, s_red[i]);
            s_red[0] = mm;
        }
        __syncthreads();
        const float gmax = s_red[0];
        __syncthreads();

        // block sum of exp(w - gmax)
        float s = __expf(w0 - gmax) + __expf(w1 - gmax) + __expf(w2 - gmax) + __expf(w3 - gmax);
#pragma unroll
        for (int off = 16; off >= 1; off >>= 1) s += __shfl_xor(s, off);
        if (lane == 0) s_red[wave] = s;
        __syncthreads();
        if (tid == 0) {
            float ss = 0.0f;
#pragma unroll
            for (int i = 0; i < NWAVES; ++i) ss += s_red[i];
            s_red[0] = ss;
        }
        __syncthreads();
        const float lse = gmax + __logf(s_red[0]);

        // clipped normalized probabilities, local inclusive scan
        float p0 = __expf(fmaxf(w0 - lse, LOG_EPS));
        float p1 = __expf(fmaxf(w1 - lse, LOG_EPS));
        float p2 = __expf(fmaxf(w2 - lse, LOG_EPS));
        float p3 = __expf(fmaxf(w3 - lse, LOG_EPS));
        float c0 = p0, c1 = c0 + p1, c2 = c1 + p2, c3 = c2 + p3;

        // warp inclusive scan of per-thread totals
        float x = c3;
#pragma unroll
        for (int off = 1; off < 32; off <<= 1) {
            float y = __shfl_up(x, off);
            if (lane >= off) x += y;
        }
        if (lane == 31) s_wave[wave] = x;
        __syncthreads();
        if (tid == 0) {
            float acc = 0.0f;
#pragma unroll
            for (int i = 0; i < NWAVES; ++i) { float v = s_wave[i]; s_wave[i] = acc; acc += v; }
            s_wave[NWAVES] = acc;
        }
        __syncthreads();
        const float base = s_wave[wave] + (x - c3);
        const float total = s_wave[NWAVES];
        s_w[i0 + 0] = base + c0;
        s_w[i0 + 1] = base + c1;
        s_w[i0 + 2] = base + c2;
        s_w[i0 + 3] = base + c3;
        __syncthreads();

        // categorical draws via inverse-CDF (branchless lower bound)
#pragma unroll
        for (int j = 0; j < 4; ++j) {
            const int jj = i0 + j;
            const float u = uniform01(hash3(0xA5117E5Du ^ b, (unsigned)t, (unsigned)jj)) * total;
            int pos = 0;
#pragma unroll
            for (int stepv = 1024; stepv >= 1; stepv >>= 1) {
                const int np = pos + stepv;
                if (np <= PF_P && s_w[np - 1] <= u) pos = np;
            }
            s_idx[jj] = (unsigned short)(pos < PF_P ? pos : PF_P - 1);
        }
        __syncthreads();

        // ================= Phase 2: gather + forward GEMM (WMMA) =================
        const __bf16* srcb = (cur == 0) ? buf0 : buf1;
        __bf16*       dstb = (cur == 0) ? buf1 : buf0;
        const size_t pbase = ((size_t)b * PF_T + t) * PF_P;

#pragma unroll 1
        for (int r = 0; r < 8; ++r) {
            const int m0 = (wave * 8 + r) * 16;
            const int arow = s_idx[m0 + nl];
            v16bf af;
            if (t == 0) {
                // gather from f32 init particles, convert to bf16 fragments
                const float* rp = init_p + (size_t)b * PF_P * PF_D + (size_t)arow * PF_D;
#pragma unroll
                for (int q = 0; q < 8; ++q) {
                    af[q]     = (__bf16)rp[koff + q];
                    af[8 + q] = (__bf16)rp[16 + koff + q];
                }
            } else {
                // bf16 state: two aligned 16B chunks per lane (global_load_b128)
                const __bf16* rp = srcb + (size_t)arow * PF_D;
                const v8bf lo = *(const v8bf*)(rp + koff);
                const v8bf hi = *(const v8bf*)(rp + 16 + koff);
                af = __builtin_shufflevector(lo, hi, 0, 1, 2, 3, 4, 5, 6, 7,
                                                     8, 9, 10, 11, 12, 13, 14, 15);
            }
            v8f acc0 = {}; v8f acc1 = {};
            acc0 = __builtin_amdgcn_wmma_f32_16x16x32_bf16(false, af, false, bA0,
                                                           (short)0, acc0, false, false);
            acc1 = __builtin_amdgcn_wmma_f32_16x16x32_bf16(false, af, false, bA1,
                                                           (short)0, acc1, false, false);
#pragma unroll
            for (int rr = 0; rr < 8; ++rr) {
                const int orow = m0 + rr + 8 * halfv;
                const float e0 = acc0[rr] + bv0 +
                    sg0 * gauss(0xB0B0u ^ b, (unsigned)(t * PF_P + orow), (unsigned)nl);
                const float e1 = acc1[rr] + bv1 +
                    sg1 * gauss(0xB0B0u ^ b, (unsigned)(t * PF_P + orow), (unsigned)(16 + nl));
                // recurrent state (bf16, L2-resident)
                dstb[(size_t)orow * PF_D + nl]      = (__bf16)e0;
                dstb[(size_t)orow * PF_D + 16 + nl] = (__bf16)e1;
                // streamed trajectory output: non-temporal (write-once, never re-read)
                __builtin_nontemporal_store(e0, &pseq[(pbase + orow) * PF_D + nl]);
                __builtin_nontemporal_store(e1, &pseq[(pbase + orow) * PF_D + 16 + nl]);
            }
        }
        __builtin_amdgcn_fence(__ATOMIC_ACQ_REL, "workgroup");
        __syncthreads();

        // ================= Phase 3: measurement GEMM (WMMA) + log-weights =================
        const float yo = obs[((size_t)b * PF_T + t) * PF_DOBS + nl];
#pragma unroll 1
        for (int r = 0; r < 8; ++r) {
            const int m0 = (wave * 8 + r) * 16;
            const __bf16* rp = dstb + (size_t)(m0 + nl) * PF_D;
            const v8bf lo = *(const v8bf*)(rp + koff);
            const v8bf hi = *(const v8bf*)(rp + 16 + koff);
            const v16bf af = __builtin_shufflevector(lo, hi, 0, 1, 2, 3, 4, 5, 6, 7,
                                                             8, 9, 10, 11, 12, 13, 14, 15);
            v8f acc = {};
            acc = __builtin_amdgcn_wmma_f32_16x16x32_bf16(false, af, false, bC,
                                                          (short)0, acc, false, false);
#pragma unroll
            for (int rr = 0; rr < 8; ++rr) {
                const float resid = (yo - acc[rr]) * rinv;
                float v = resid * resid;
                v += __shfl_xor(v, 1);
                v += __shfl_xor(v, 2);
                v += __shfl_xor(v, 4);
                v += __shfl_xor(v, 8);   // sum over 16 obs dims within half-wave
                if (nl == rr) {
                    const int orow = m0 + rr + 8 * halfv;
                    const float lw = -0.5f * v + const_term;
                    s_w[orow] = lw;
                    __builtin_nontemporal_store(lw, &wseq[pbase + orow]);
                }
            }
        }
        __syncthreads();

        cur = (t == 0) ? 1 : (1 - cur);   // after t==0 the state lives in buf1
    }
}

extern "C" void kernel_launch(void* const* d_in, const int* in_sizes, int n_in,
                              void* d_out, int out_size, void* d_ws, size_t ws_size,
                              hipStream_t stream) {
    (void)in_sizes; (void)n_in; (void)out_size; (void)ws_size;
    const float* obs       = (const float*)d_in[0];
    const float* init_p    = (const float*)d_in[1];
    const float* Amat      = (const float*)d_in[2];
    const float* bvec      = (const float*)d_in[3];
    const float* log_sigma = (const float*)d_in[4];
    const float* Cmat      = (const float*)d_in[5];
    const float* log_r     = (const float*)d_in[6];

    float* pseq = (float*)d_out;
    float* wseq = pseq + (size_t)PF_B * PF_T * PF_P * PF_D;
    __bf16* ws  = (__bf16*)d_ws;

    pf_persistent_kernel<<<PF_B, NTHREADS, 0, stream>>>(
        obs, init_p, Amat, bvec, log_sigma, Cmat, log_r, pseq, wseq, ws);
}
